// IoUHead_81037442941243
// MI455X (gfx1250) — matrix operand order repaired and checked by reading, not compiled, not run on hardware
//
#include <hip/hip_runtime.h>
#include <hip/hip_bf16.h>
#include <math.h>

// ---------------- types / constants ----------------
typedef __bf16 bf16_t;
typedef __attribute__((ext_vector_type(16))) __bf16       v16bf;
typedef __attribute__((ext_vector_type(8)))  float        v8f;
typedef __attribute__((ext_vector_type(4)))  unsigned int u32x4;
typedef __attribute__((ext_vector_type(8)))  int          i32x8;
typedef __attribute__((ext_vector_type(4)))  int          i32x4;

#define BS      2
#define NPOINT  128
#define NPTS    8192
#define CFEAT   128
#define KGRID   64
#define MCOR    (NPOINT * KGRID)   // 8192 corners per batch
#define KPAD    160                // 131 -> 160 (5 chunks of 32)
#define MLP1_O  128
#define MLP2_O  256
#define IOUH    128
#define NT_PER  4                  // N-tiles per wave in WMMA GEMMs
#define PI_F    3.14159265358979323846f

__device__ __forceinline__ unsigned short bf16bits(float f) {
  union { float f; unsigned u; } v; v.f = f;
  unsigned r = v.u + 0x7FFFu + ((v.u >> 16) & 1u);   // round-to-nearest-even
  return (unsigned short)(r >> 16);
}
__device__ __forceinline__ bf16_t to_bf16(float f) {
  union { unsigned short s; bf16_t b; } o; o.s = bf16bits(f);
  return o.b;
}
__device__ __forceinline__ unsigned pack2(float a, float b) {
  return (unsigned)bf16bits(a) | ((unsigned)bf16bits(b) << 16);
}

// ---- Tensor Data Mover: DMA `nfloats` f32 from global -> LDS (2D D#, 1 row) ----
// 6-arg builtin variant: (u32x4 g0, i32x8 g1, i32x4 g2, i32x4 g3, i32x8 extra, i32 cpol)
__device__ __forceinline__ void tdm_load_f32(const float* gsrc, unsigned lds_off, int nfloats) {
  unsigned long long ga = (unsigned long long)(uintptr_t)gsrc;
  u32x4 g0;
  g0[0] = 1u;                                            // count=1, is_restore=0, gather off
  g0[1] = lds_off;                                       // lds_addr (bytes)
  g0[2] = (unsigned)(ga & 0xFFFFFFFFu);                  // global_addr[31:0]
  g0[3] = (unsigned)((ga >> 32) & 0x01FFFFFFu) | (2u << 30);  // global_addr[56:32] | type=2
  unsigned nf = (unsigned)nfloats;
  i32x8 g1;
  g1[0] = (int)(2u << 16);                               // wg_mask=0, data_size=2 (4 bytes)
  g1[1] = (int)((nf & 0xFFFFu) << 16);                   // tensor_dim0[15:0]
  g1[2] = (int)(((nf >> 16) & 0xFFFFu) | (1u << 16));    // tensor_dim0[31:16] | tensor_dim1=1
  g1[3] = (int)((nf & 0xFFFFu) << 16);                   // tensor_dim1 hi(0)  | tile_dim0
  g1[4] = 1;                                             // tile_dim1=1, tile_dim2=0
  g1[5] = (int)nf;                                       // tensor_dim0_stride[31:0]
  g1[6] = (int)((nf & 0xFFFFu) << 16);                   // s0[47:32]=0 | tensor_dim1_stride lo
  g1[7] = 0;
  i32x4 z4 = { 0, 0, 0, 0 };
  i32x8 z8 = { 0, 0, 0, 0, 0, 0, 0, 0 };
  __builtin_amdgcn_tensor_load_to_lds(g0, g1, z4, z4, z8, 0);
}

// ---------------- 1. decode boxes + build rel / corners ----------------
__global__ void k_decode(const float* __restrict__ cand_xyz,
                         const float* __restrict__ offsets,
                         const float* __restrict__ angle_cls,
                         const float* __restrict__ angle_res,
                         float* __restrict__ rel,
                         float* __restrict__ cor) {
  const int box = blockIdx.x;            // b*NPOINT + p
  const int b   = box >> 7;
  const int p   = box & 127;
  const int k   = threadIdx.x;           // 0..63

  const float* ac = angle_cls + box * 12;
  int   best = 0; float bv = ac[0];
  #pragma unroll
  for (int i = 1; i < 12; ++i) { float v = ac[i]; if (v > bv) { bv = v; best = i; } }
  float ares    = angle_res[box * 12 + best];
  float angle   = (float)best * (2.0f * PI_F / 12.0f) + ares;
  float heading = (angle > PI_F) ? (angle - 2.0f * PI_F) : angle;

  float cx = cand_xyz[box * 3 + 0] + offsets[box * 6 + 0];
  float cy = cand_xyz[box * 3 + 1] + offsets[box * 6 + 1];
  float cz = cand_xyz[box * 3 + 2] + offsets[box * 6 + 2];
  float lx = fmaxf(offsets[box * 6 + 3] * 2.0f, 0.1f);
  float ly = fmaxf(offsets[box * 6 + 4] * 2.0f, 0.1f);
  float lz = fmaxf(offsets[box * 6 + 5] * 2.0f, 0.1f);

  const float gs[4] = { -1.0f, -1.0f / 3.0f, 1.0f / 3.0f, 1.0f };
  int ix = k >> 4, iy = (k >> 2) & 3, iz = k & 3;
  float wx = gs[ix] * lx, wy = gs[iy] * ly, wz = gs[iz] * lz;

  float c = __cosf(heading), s = __sinf(heading);
  float rx = wx * c - wy * s;
  float ry = wx * s + wy * c;
  float rz = wz;

  const size_t m = (size_t)b * MCOR + (size_t)p * KGRID + k;
  rel[m * 3 + 0] = rx; rel[m * 3 + 1] = ry; rel[m * 3 + 2] = rz;
  cor[m * 3 + 0] = rx + cx; cor[m * 3 + 1] = ry + cy; cor[m * 3 + 2] = rz + cz;
}

// ---------------- 2. three_nn: TDM-staged point tiles, per-thread top-3 ----------------
// grid: (MCOR/256, BS) x 256 threads
__global__ void k_three_nn(const float* __restrict__ xyz,   // [BS,NPTS,3]
                           const float* __restrict__ cor,   // [BS,MCOR,3]
                           int*   __restrict__ nn_idx,      // [BS,MCOR,3]
                           float* __restrict__ nn_w) {      // [BS,MCOR,3]
  __shared__ float sbuf[768];            // 256 points x 3 (interleaved xyz)
  const int tid = threadIdx.x;
  const int b   = blockIdx.y;
  const int m   = blockIdx.x * 256 + tid;

  const size_t cbase = ((size_t)b * MCOR + m) * 3;
  const float cx = cor[cbase + 0];
  const float cy = cor[cbase + 1];
  const float cz = cor[cbase + 2];

  float d0 = 3.0e38f, d1 = 3.0e38f, d2 = 3.0e38f;
  int   i0 = 0, i1 = 0, i2 = 0;

  const float* PB = xyz + (size_t)b * NPTS * 3;
  const unsigned lds_off = (unsigned)(uintptr_t)(void*)sbuf;

  for (int chunk = 0; chunk < NPTS / 256; ++chunk) {
    __syncthreads();                      // previous chunk fully consumed
    if (tid < 32) {                       // one wave drives the TDM
      tdm_load_f32(PB + (size_t)chunk * 768, lds_off, 768);
      __builtin_amdgcn_s_wait_tensorcnt(0);
    }
    __syncthreads();                      // data visible to whole block

    const int pbase = chunk * 256;
    #pragma unroll 8
    for (int i = 0; i < 256; ++i) {
      float dx = cx - sbuf[3 * i + 0];
      float dy = cy - sbuf[3 * i + 1];
      float dz = cz - sbuf[3 * i + 2];
      float d  = __builtin_fmaf(dx, dx, __builtin_fmaf(dy, dy, dz * dz));
      if (d < d2) {
        int pid = pbase + i;
        if (d < d1) {
          d2 = d1; i2 = i1;
          if (d < d0) { d1 = d0; i1 = i0; d0 = d; i0 = pid; }
          else        { d1 = d;  i1 = pid; }
        } else { d2 = d; i2 = pid; }
      }
    }
  }

  float w0 = 1.0f / (sqrtf(d0) + 1e-8f);
  float w1 = 1.0f / (sqrtf(d1) + 1e-8f);
  float w2 = 1.0f / (sqrtf(d2) + 1e-8f);
  float inv = 1.0f / (w0 + w1 + w2);
  const size_t o = ((size_t)b * MCOR + m) * 3;
  nn_idx[o + 0] = i0; nn_idx[o + 1] = i1; nn_idx[o + 2] = i2;
  nn_w[o + 0] = w0 * inv; nn_w[o + 1] = w1 * inv; nn_w[o + 2] = w2 * inv;
}

// ---------------- 3. interpolate -> bf16 corner-major activations ----------------
// featB[b][n][KPAD]: k<3 rel, 3..130 interp, >=131 zero pad. grid: BS*MCOR blocks x KPAD thr
__global__ void k_interp(const float* __restrict__ feat,   // [BS,CFEAT,NPTS]
                         const float* __restrict__ rel,    // [BS,MCOR,3]
                         const int*   __restrict__ nn_idx,
                         const float* __restrict__ nn_w,
                         bf16_t* __restrict__ featB) {     // [BS,MCOR,KPAD]
  const int bm = blockIdx.x;          // b*MCOR + m
  const int b  = bm >> 13;
  const int m  = bm & (MCOR - 1);
  const int k  = threadIdx.x;         // 0..159

  bf16_t* out = featB + (size_t)bm * KPAD + k;
  if (k < 3) {
    *out = to_bf16(rel[((size_t)b * MCOR + m) * 3 + k]);
  } else if (k < 131) {
    const size_t o = ((size_t)b * MCOR + m) * 3;
    const int   c  = k - 3;
    const float* F = feat + ((size_t)b * CFEAT + c) * NPTS;
    float v = nn_w[o + 0] * F[nn_idx[o + 0]]
            + nn_w[o + 1] * F[nn_idx[o + 1]]
            + nn_w[o + 2] * F[nn_idx[o + 2]];
    *out = to_bf16(v);
  } else {
    *out = to_bf16(0.0f);
  }
}

// ---------------- 4. pack weights into WMMA A-fragment order (bf16) ----------------
// A 16x32 bf16: lane l (row=l&15, half h=l>>4), elem j: K = j + 16*h
__global__ void k_packA(const float* __restrict__ W, bf16_t* __restrict__ pack,
                        int Kcols, int MT, int KC) {
  int t = blockIdx.x * blockDim.x + threadIdx.x;
  if (t >= MT * KC * 32) return;
  int lane = t & 31;
  int frag = t >> 5;
  int kc   = frag % KC;
  int mt   = frag / KC;
  int mr   = mt * 16 + (lane & 15);
  int h    = lane >> 4;
  bf16_t* dst = pack + ((size_t)frag * 32 + lane) * 16;
  #pragma unroll
  for (int j = 0; j < 16; ++j) {
    int kk = kc * 32 + j + 16 * h;
    float v = (kk < Kcols) ? W[(size_t)mr * Kcols + kk] : 0.0f;
    dst[j] = to_bf16(v);
  }
}

// ---------------- 5. MLP1: h1 = relu(W1 @ feat + b1), bf16 WMMA ----------------
// grid: (MCOR/(16*NT_PER), BS) x 256; wave = M-tile, sweeps NT_PER N-tiles reusing A
__global__ void k_gemm1(const bf16_t* __restrict__ featB, const bf16_t* __restrict__ w1p,
                        const float* __restrict__ b1, bf16_t* __restrict__ h1B) {
  const int lane = threadIdx.x & 31;
  const int mt   = threadIdx.x >> 5;     // 0..7
  const int b    = blockIdx.y;
  const int h    = lane >> 4;
  const int nl   = lane & 15;
  const bf16_t* Bb = featB + (size_t)b * MCOR * KPAD;

  v16bf a[5];
  #pragma unroll
  for (int kc = 0; kc < 5; ++kc)
    a[kc] = *(const v16bf*)(w1p + ((size_t)(mt * 5 + kc) * 32 + lane) * 16);

  const int c0 = mt * 16 + 8 * h;        // first of 8 contiguous output channels
  float bias[8];
  #pragma unroll
  for (int r = 0; r < 8; ++r) bias[r] = b1[c0 + r];

  for (int t = 0; t < NT_PER; ++t) {
    const int n = (blockIdx.x * NT_PER + t) * 16 + nl;
    const bf16_t* bp = Bb + (size_t)n * KPAD + h * 16;
    if (t + 1 < NT_PER) __builtin_prefetch(bp + 16 * KPAD, 0, 1);
    v8f acc = { 0.f, 0.f, 0.f, 0.f, 0.f, 0.f, 0.f, 0.f };
    #pragma unroll
    for (int kc = 0; kc < 5; ++kc) {
      v16bf bf = *(const v16bf*)(bp + kc * 32);
      acc = __builtin_amdgcn_wmma_f32_16x16x32_bf16(false, a[kc], false, bf,
                                                    (short)0, acc, false, false);
    }
    u32x4 st;
    #pragma unroll
    for (int r = 0; r < 4; ++r)
      st[r] = pack2(fmaxf(acc[2 * r] + bias[2 * r], 0.0f),
                    fmaxf(acc[2 * r + 1] + bias[2 * r + 1], 0.0f));
    *(u32x4*)(h1B + ((size_t)b * MCOR + n) * MLP1_O + c0) = st;
  }
}

// ---------------- 6. MLP2: h2 = relu(W2 @ h1 + b2), bf16 WMMA, f32 out ----------------
__global__ void k_gemm2(const bf16_t* __restrict__ h1B, const bf16_t* __restrict__ w2p,
                        const float* __restrict__ b2, float* __restrict__ h2) {
  const int lane = threadIdx.x & 31;
  const int wave = threadIdx.x >> 5;     // 0..7
  const int b    = blockIdx.y;
  const int h    = lane >> 4;
  const int nl   = lane & 15;
  const bf16_t* Bb = h1B + (size_t)b * MCOR * MLP1_O;

  for (int mi = 0; mi < 2; ++mi) {
    const int mt = wave + mi * 8;        // 0..15
    v16bf a[4];
    #pragma unroll
    for (int kc = 0; kc < 4; ++kc)
      a[kc] = *(const v16bf*)(w2p + ((size_t)(mt * 4 + kc) * 32 + lane) * 16);
    const int c0 = mt * 16 + 8 * h;
    float bias[8];
    #pragma unroll
    for (int r = 0; r < 8; ++r) bias[r] = b2[c0 + r];

    for (int t = 0; t < NT_PER; ++t) {
      const int n = (blockIdx.x * NT_PER + t) * 16 + nl;
      const bf16_t* bp = Bb + (size_t)n * MLP1_O + h * 16;
      if (t + 1 < NT_PER) __builtin_prefetch(bp + 16 * MLP1_O, 0, 1);
      v8f acc = { 0.f, 0.f, 0.f, 0.f, 0.f, 0.f, 0.f, 0.f };
      #pragma unroll
      for (int kc = 0; kc < 4; ++kc) {
        v16bf bf = *(const v16bf*)(bp + kc * 32);
        acc = __builtin_amdgcn_wmma_f32_16x16x32_bf16(false, a[kc], false, bf,
                                                      (short)0, acc, false, false);
      }
      #pragma unroll
      for (int r = 0; r < 8; ++r)
        h2[((size_t)b * MLP2_O + c0 + r) * MCOR + n] = fmaxf(acc[r] + bias[r], 0.0f);
    }
  }
}

// ---------------- 7. max over K=64 grid points ----------------
__global__ void k_maxk(const float* __restrict__ h2, float* __restrict__ h2max) {
  int t = blockIdx.x * blockDim.x + threadIdx.x;        // BS*256*128 threads
  if (t >= BS * MLP2_O * NPOINT) return;
  int p  = t & 127;
  int ch = (t >> 7) & 255;
  int b  = t >> 15;
  const float* src = h2 + ((size_t)b * MLP2_O + ch) * MCOR + (size_t)p * KGRID;
  float mx = src[0];
  #pragma unroll 8
  for (int k = 1; k < KGRID; ++k) mx = fmaxf(mx, src[k]);
  h2max[((size_t)b * MLP2_O + ch) * NPOINT + p] = mx;
}

// ---------------- 8. small dense layers of IoU head ----------------
__global__ void k_gemm_small(const float* __restrict__ in, const float* __restrict__ W,
                             const float* __restrict__ bias, float* __restrict__ out,
                             int O, int Cdim) {
  int t = blockIdx.x * blockDim.x + threadIdx.x;        // BS*O*128 threads
  if (t >= BS * O * NPOINT) return;
  int p = t & 127;
  int o = (t >> 7) % O;
  int b = t / (O * NPOINT);
  const float* x = in + ((size_t)b * Cdim) * NPOINT + p;
  const float* w = W + (size_t)o * Cdim;
  float acc = bias[o];
  for (int c = 0; c < Cdim; ++c) acc = __builtin_fmaf(w[c], x[(size_t)c * NPOINT], acc);
  out[((size_t)b * O + o) * NPOINT + p] = acc;
}

// training-mode BN over (batch, length) + relu. One block per channel, 256 threads
__global__ void k_bn_relu(const float* __restrict__ y, const float* __restrict__ g,
                          const float* __restrict__ be, float* __restrict__ out, int C) {
  __shared__ float ssum[256];
  __shared__ float ssq[256];
  const int ch = blockIdx.x;
  const int t  = threadIdx.x;          // b = t>>7, p = t&127
  const int b  = t >> 7, p = t & 127;
  float v = y[((size_t)b * C + ch) * NPOINT + p];
  ssum[t] = v; ssq[t] = v * v;
  __syncthreads();
  for (int s = 128; s > 0; s >>= 1) {
    if (t < s) { ssum[t] += ssum[t + s]; ssq[t] += ssq[t + s]; }
    __syncthreads();
  }
  float mean = ssum[0] * (1.0f / 256.0f);
  float var  = ssq[0] * (1.0f / 256.0f) - mean * mean;
  float inv  = rsqrtf(var + 1e-5f);
  float o = fmaxf((v - mean) * inv * g[ch] + be[ch], 0.0f);
  out[((size_t)b * C + ch) * NPOINT + p] = o;
}

// ---------------- launcher ----------------
extern "C" void kernel_launch(void* const* d_in, const int* in_sizes, int n_in,
                              void* d_out, int out_size, void* d_ws, size_t ws_size,
                              hipStream_t stream) {
  const float* original_xyz   = (const float*)d_in[0];
  const float* original_feat  = (const float*)d_in[1];
  const float* candidate_xyz  = (const float*)d_in[2];
  // d_in[3] = pred_cls (unused by forward)
  const float* pred_offsets   = (const float*)d_in[4];
  const float* pred_angle_cls = (const float*)d_in[5];
  const float* pred_angle_res = (const float*)d_in[6];
  const float* w_mlp1 = (const float*)d_in[7];
  const float* b_mlp1 = (const float*)d_in[8];
  const float* w_mlp2 = (const float*)d_in[9];
  const float* b_mlp2 = (const float*)d_in[10];
  const float* w_iou0 = (const float*)d_in[11];
  const float* b_iou0 = (const float*)d_in[12];
  const float* g_iou0 = (const float*)d_in[13];
  const float* be_iou0= (const float*)d_in[14];
  const float* w_iou1 = (const float*)d_in[15];
  const float* b_iou1 = (const float*)d_in[16];
  const float* g_iou1 = (const float*)d_in[17];
  const float* be_iou1= (const float*)d_in[18];
  const float* w_iou2 = (const float*)d_in[19];
  const float* b_iou2 = (const float*)d_in[20];
  float* out = (float*)d_out;

  // ---- workspace layout (256B aligned offsets) ----
  char* ws = (char*)d_ws;
  size_t off = 0;
  auto alloc = [&](size_t bytes) { void* p = ws + off; off = (off + bytes + 255) & ~(size_t)255; return p; };
  float*  rel    = (float*) alloc((size_t)BS * MCOR * 3 * 4);
  float*  cor    = (float*) alloc((size_t)BS * MCOR * 3 * 4);
  int*    nn_idx = (int*)   alloc((size_t)BS * MCOR * 3 * 4);
  float*  nn_w   = (float*) alloc((size_t)BS * MCOR * 3 * 4);
  bf16_t* featB  = (bf16_t*)alloc((size_t)BS * MCOR * KPAD * 2);
  bf16_t* w1p    = (bf16_t*)alloc((size_t)8  * 5 * 512 * 2);
  bf16_t* w2p    = (bf16_t*)alloc((size_t)16 * 4 * 512 * 2);
  bf16_t* h1B    = (bf16_t*)alloc((size_t)BS * MCOR * MLP1_O * 2);
  float*  h2     = (float*) alloc((size_t)BS * MLP2_O * MCOR * 4);
  float*  h2max  = (float*) alloc((size_t)BS * MLP2_O * NPOINT * 4);
  float*  y0     = (float*) alloc((size_t)BS * IOUH * NPOINT * 4);
  float*  y0n    = (float*) alloc((size_t)BS * IOUH * NPOINT * 4);
  float*  y1     = (float*) alloc((size_t)BS * IOUH * NPOINT * 4);
  float*  y1n    = (float*) alloc((size_t)BS * IOUH * NPOINT * 4);
  (void)ws_size; (void)in_sizes; (void)n_in; (void)out_size;

  // 1. decode + corners
  k_decode<<<dim3(BS * NPOINT), dim3(KGRID), 0, stream>>>(
      candidate_xyz, pred_offsets, pred_angle_cls, pred_angle_res, rel, cor);

  // 2. three_nn (TDM-staged)
  k_three_nn<<<dim3(MCOR / 256, BS), dim3(256), 0, stream>>>(original_xyz, cor, nn_idx, nn_w);

  // 3. interpolation -> bf16 activations (corner-major)
  k_interp<<<dim3(BS * MCOR), dim3(KPAD), 0, stream>>>(original_feat, rel, nn_idx, nn_w, featB);

  // 4. weight packing for WMMA
  k_packA<<<dim3((8 * 5 * 32 + 255) / 256), dim3(256), 0, stream>>>(w_mlp1, w1p, 131, 8, 5);
  k_packA<<<dim3((16 * 4 * 32 + 255) / 256), dim3(256), 0, stream>>>(w_mlp2, w2p, 128, 16, 4);

  // 5/6. shared-MLP GEMMs on WMMA
  k_gemm1<<<dim3(MCOR / (16 * NT_PER), BS), dim3(256), 0, stream>>>(featB, w1p, b_mlp1, h1B);
  k_gemm2<<<dim3(MCOR / (16 * NT_PER), BS), dim3(256), 0, stream>>>(h1B, w2p, b_mlp2, h2);

  // 7. max-pool over grid points
  k_maxk<<<dim3((BS * MLP2_O * NPOINT + 255) / 256), dim3(256), 0, stream>>>(h2, h2max);

  // 8. IoU head: fc -> bn+relu -> fc -> bn+relu -> fc
  k_gemm_small<<<dim3((BS * IOUH * NPOINT + 255) / 256), dim3(256), 0, stream>>>(
      h2max, w_iou0, b_iou0, y0, IOUH, MLP2_O);
  k_bn_relu<<<dim3(IOUH), dim3(256), 0, stream>>>(y0, g_iou0, be_iou0, y0n, IOUH);
  k_gemm_small<<<dim3((BS * IOUH * NPOINT + 255) / 256), dim3(256), 0, stream>>>(
      y0n, w_iou1, b_iou1, y1, IOUH, IOUH);
  k_bn_relu<<<dim3(IOUH), dim3(256), 0, stream>>>(y1, g_iou1, be_iou1, y1n, IOUH);
  k_gemm_small<<<dim3((BS * 1 * NPOINT + 255) / 256), dim3(256), 0, stream>>>(
      y1n, w_iou2, b_iou2, out, 1, IOUH);
}
